// SpatialInformerLayer_56504589746722
// MI455X (gfx1250) — compile-verified, matrix operand order checked by reading, not run on hardware
//
#include <hip/hip_runtime.h>

// ---------------------------------------------------------------------------
// SpatialInformerLayer for MI455X (gfx1250): bf16 WMMA everywhere, with
// TDM (tensor_load_to_lds) staging of shared A panels + global prefetch.
// B=8, N=1024, D_MODEL=512, H=8, E=64, D_FF=2048, u = 25% of 1024 = 256.
// ---------------------------------------------------------------------------

#define B_   8
#define N_   1024
#define DM_  512
#define H_   8
#define E_   64
#define DFF_ 2048
#define MR_  (B_ * N_)     // 8192 token rows
#define BH_  (B_ * H_)     // 64
#define U_   256           // top-u queries (complexity = 25)

typedef __bf16 bf16_t;
typedef __attribute__((ext_vector_type(16))) __bf16 v16bf;
typedef __attribute__((ext_vector_type(8)))  float  v8f;
typedef __attribute__((ext_vector_type(4)))  unsigned tdm_v4u;
typedef __attribute__((ext_vector_type(8)))  int      tdm_v8i;
typedef __attribute__((ext_vector_type(4)))  int      tdm_v4i;

__device__ __forceinline__ bf16_t f2bf(float f) {
  unsigned u = __builtin_bit_cast(unsigned, f);
  unsigned r = u + 0x7FFFu + ((u >> 16) & 1u);   // round-to-nearest-even
  unsigned short h = (unsigned short)(r >> 16);
  return __builtin_bit_cast(bf16_t, h);
}
__device__ __forceinline__ float bf2f(bf16_t b) {
  unsigned u = ((unsigned)__builtin_bit_cast(unsigned short, b)) << 16;
  return __builtin_bit_cast(float, u);
}

// CDNA5 16-bit A/B operand tile, row-major source with leading dim `ld`:
// lane<16: row=lane,   K 0..7 in elems 0..7, K 16..23 in elems 8..15
// lane>=16: row=lane-16, K 8..15 / 24..31.
__device__ __forceinline__ v16bf load_tile_g(const bf16_t* __restrict__ base,
                                             int ld, int lane) {
  const bf16_t* p = base + (lane & 15) * ld + ((lane >> 4) << 3);
  v16bf r;
#pragma unroll
  for (int i = 0; i < 8; ++i) { r[i] = p[i]; r[i + 8] = p[i + 16]; }
  return r;
}

__device__ __forceinline__ v8f wmma_bf16(v16bf a, v16bf b, v8f c) {
  return __builtin_amdgcn_wmma_f32_16x16x32_bf16(false, a, false, b,
                                                 (short)0, c, false, false);
}

// ---------------------------------------------------------------------------
// Stage a contiguous [rows x cols] bf16 panel (row stride == cols) from
// global into LDS.  Preferred path: one Tensor-Data-Mover descriptor issued
// by wave 0 (D# per CDNA5 ISA), completion via
// TENSORcnt; fallback: cooperative copy.  Ends with a workgroup barrier.
// ---------------------------------------------------------------------------
__device__ __forceinline__ void stage_panel(const bf16_t* __restrict__ g,
                                            bf16_t* sh, int rows, int cols,
                                            int wave) {
#if defined(__gfx1250__) && __has_builtin(__builtin_amdgcn_tensor_load_to_lds)
  if (wave == 0) {
    unsigned long long ga = (unsigned long long)(size_t)g;
    unsigned lds_off = (unsigned)(size_t)sh;   // low 32 bits = LDS byte addr
    tdm_v4u g0;
    g0[0] = 1u;                                 // count=1, user descriptor
    g0[1] = lds_off;                            // lds_addr [63:32]
    g0[2] = (unsigned)(ga & 0xffffffffu);       // global_addr [95:64]
    g0[3] = (unsigned)((ga >> 32) & 0x1ffffffu) // global_addr [120:96]
          | (2u << 30);                         // type=2 ("image") [127:126]
    tdm_v8i g1;
    g1[0] = (int)(1u << 16);                    // wg_mask=0, data_size=1 (2B)
    g1[1] = (int)(((unsigned)cols & 0xffffu) << 16);          // tdim0 lo16
    g1[2] = (int)((((unsigned)cols >> 16) & 0xffffu)          // tdim0 hi16
          | (((unsigned)rows & 0xffffu) << 16));              // tdim1 lo16
    g1[3] = (int)(((unsigned)cols & 0xffffu) << 16);          // tile_dim0
    g1[4] = (int)((unsigned)rows & 0xffffu);                  // tile_dim1
    g1[5] = (int)(unsigned)cols;                // tensor_dim0_stride [191:160]
    g1[6] = 0;
    g1[7] = 0;
    tdm_v4i z4 = {0, 0, 0, 0};
#if __has_include(<hip/amd_detail/amd_gfx1250_TDM.h>)
    tdm_v8i z8 = {0, 0, 0, 0, 0, 0, 0, 0};      // 6-arg toolchain variant
    __builtin_amdgcn_tensor_load_to_lds(g0, g1, z4, z4, z8, 0);
#else
    __builtin_amdgcn_tensor_load_to_lds(g0, g1, z4, z4, 0);   // 5-arg (ROCm)
#endif
#if __has_builtin(__builtin_amdgcn_s_wait_tensorcnt)
    __builtin_amdgcn_s_wait_tensorcnt(0);
#endif
  }
#else
  (void)wave;
  for (int i = threadIdx.x; i < rows * cols; i += blockDim.x) sh[i] = g[i];
#endif
  __syncthreads();
}

// ---------------------------------------------------------------------------
// Kernel: fp32 -> bf16 conversion
// ---------------------------------------------------------------------------
__global__ void k_f32_to_bf16(const float* __restrict__ src,
                              bf16_t* __restrict__ dst, int n) {
  int i = blockIdx.x * blockDim.x + threadIdx.x;
  if (i < n) dst[i] = f2bf(src[i]);
}

// ---------------------------------------------------------------------------
// Kernel: strip GEMM  C = A[M,K] * W[Nc,K]^T + bias.  8 waves per block; all
// 8 strips of a block share one 16-row A panel -> TDM-stage it in LDS once.
// mode 0: write head-split [B,H,N,E]; mode 2: head-split transposed [B,H,E,N];
// mode 3: flat [M,Nc] with ReLU.
// ---------------------------------------------------------------------------
__global__ void k_gemm_strip(const bf16_t* __restrict__ A,
                             const bf16_t* __restrict__ W,
                             const float* __restrict__ bias,
                             int Mrows, int Ncols, int K, int mode,
                             bf16_t* __restrict__ out) {
  extern __shared__ char smem[];
  bf16_t* shA = (bf16_t*)smem;                  // [16][K]
  int lane  = threadIdx.x & 31;
  int wave  = threadIdx.x >> 5;
  int strip = blockIdx.x * (blockDim.x >> 5) + wave;
  int nstrip = Ncols >> 6;
  int m0 = (strip / nstrip) << 4;               // block-uniform (8 | nstrip)
  int c0 = (strip % nstrip) << 6;

  stage_panel(A + m0 * K, shA, 16, K, wave);
  if (m0 >= Mrows) return;                      // wave-uniform

  v8f acc[4] = {};
  for (int kk = 0; kk < K; kk += 32) {
#pragma unroll
    for (int c = 0; c < 4; ++c)                 // prefetch next W K-step
      __builtin_prefetch((const void*)(W + (c0 + (c << 4)) * K + kk + 32), 0, 1);
    v16bf a = load_tile_g(shA + kk, K, lane);   // DS loads from staged panel
#pragma unroll
    for (int c = 0; c < 4; ++c) {
      v16bf b = load_tile_g(W + (c0 + (c << 4)) * K + kk, K, lane);
      acc[c] = wmma_bf16(a, b, acc[c]);
    }
  }
  int colbase = lane & 15;
  int rowoff  = (lane >> 4) << 3;
#pragma unroll
  for (int c = 0; c < 4; ++c) {
#pragma unroll
    for (int i = 0; i < 8; ++i) {
      int row = m0 + rowoff + i;
      int col = c0 + (c << 4) + colbase;
      float v = acc[c][i] + bias[col];
      if (mode == 3) {
        out[row * Ncols + col] = f2bf(v > 0.f ? v : 0.f);
      } else {
        int b_ = row >> 10, n_ = row & (N_ - 1);
        int h_ = col >> 6,  e_ = col & (E_ - 1);
        int bh = b_ * H_ + h_;
        if (mode == 2) out[(bh * E_ + e_) * N_ + n_] = f2bf(v);
        else           out[(bh * N_ + n_) * E_ + e_] = f2bf(v);
      }
    }
  }
}

// ---------------------------------------------------------------------------
// Kernel: QK^T for 16 queries x all 1024 keys (scores to LDS), then the
// ProbSparse measure M = max(QKs) - mean(QKs) via index_sample gather.
// grid (BH, N/16), 128 threads (4 waves, each covers 256 keys).
// ---------------------------------------------------------------------------
__global__ void k_scores_M(const bf16_t* __restrict__ qg,
                           const bf16_t* __restrict__ kg,
                           const int* __restrict__ idx,
                           float* __restrict__ Mout) {
  extern __shared__ char smem[];
  float* sc  = (float*)smem;       // [16][1024]
  float* red = sc + 16 * N_;       // [256]
  int bh = blockIdx.x;
  int q0 = blockIdx.y << 4;
  int lane = threadIdx.x & 31, wave = threadIdx.x >> 5;

  const bf16_t* qbase = qg + (bh * N_ + q0) * E_;
  v16bf a0 = load_tile_g(qbase,      E_, lane);
  v16bf a1 = load_tile_g(qbase + 32, E_, lane);
  int colbase = lane & 15, rowoff = (lane >> 4) << 3;

  for (int t = 0; t < 16; ++t) {
    int kr = (wave << 8) + (t << 4);
    const bf16_t* kbase = kg + (bh * N_ + kr) * E_;
    v8f acc = {};
    acc = wmma_bf16(a0, load_tile_g(kbase,      E_, lane), acc);
    acc = wmma_bf16(a1, load_tile_g(kbase + 32, E_, lane), acc);
#pragma unroll
    for (int i = 0; i < 8; ++i)
      sc[(rowoff + i) * N_ + kr + colbase] = acc[i];
  }
  __syncthreads();

  int row = threadIdx.x >> 3;   // 0..15
  int sub = threadIdx.x & 7;
  const int* irow = idx + (q0 + row) * N_;
  float mx = -3.4e38f, sum = 0.f;
  for (int j = sub; j < N_; j += 8) {
    float v = sc[row * N_ + irow[j]];
    mx = fmaxf(mx, v);
    sum += v;
  }
  red[row * 8 + sub]       = mx;
  red[128 + row * 8 + sub] = sum;
  __syncthreads();
  if (sub == 0) {
    float m2 = -3.4e38f, s2 = 0.f;
    for (int j = 0; j < 8; ++j) {
      m2 = fmaxf(m2, red[row * 8 + j]);
      s2 += red[128 + row * 8 + j];
    }
    Mout[bh * N_ + q0 + row] = m2 - s2 * (1.f / (float)N_);
  }
}

// ---------------------------------------------------------------------------
// Kernel: stable top-U by rank counting (matches lax.top_k ordering:
// descending values, ties broken by smaller index).  grid BH, 256 threads.
// ---------------------------------------------------------------------------
__global__ void k_topk(const float* __restrict__ Mm, int* __restrict__ topidx) {
  __shared__ float mrow[N_];
  int bh = blockIdx.x;
  for (int i = threadIdx.x; i < N_; i += blockDim.x) mrow[i] = Mm[bh * N_ + i];
  __syncthreads();
  for (int i = threadIdx.x; i < N_; i += blockDim.x) {
    float mi = mrow[i];
    int rank = 0;
    for (int j = 0; j < N_; ++j) {
      float mj = mrow[j];
      rank += (mj > mi) || (mj == mi && j < i);
    }
    if (rank < U_) topidx[bh * U_ + rank] = i;
  }
}

// ---------------------------------------------------------------------------
// Kernel: mean of V over sequence (uses transposed V layout).
// ---------------------------------------------------------------------------
__global__ void k_vmean(const bf16_t* __restrict__ vT, float* __restrict__ vmean) {
  int bh = blockIdx.x;        // 64
  int e  = threadIdx.x;       // 64
  const bf16_t* p = vT + (bh * E_ + e) * N_;
  float s = 0.f;
  for (int n = 0; n < N_; ++n) s += bf2f(p[n]);
  vmean[bh * E_ + e] = s * (1.f / (float)N_);
}

// Fill attention-output buffer [B*N, 512] with broadcast vmean.
__global__ void k_ctx_init(const float* __restrict__ vmean,
                           bf16_t* __restrict__ ctx) {
  int i = blockIdx.x * blockDim.x + threadIdx.x;
  if (i >= MR_ * DM_) return;
  int col = i & (DM_ - 1);
  int row = i >> 9;
  int b_ = row >> 10;
  int h_ = col >> 6, e_ = col & (E_ - 1);
  ctx[i] = f2bf(vmean[(b_ * H_ + h_) * E_ + e_]);
}

// ---------------------------------------------------------------------------
// Kernel: attention for 16 selected queries: recompute scores (WMMA),
// softmax in LDS, P @ V^T (WMMA, K=1024), scatter rows into ctx.
// grid (BH, U/16), 128 threads (4 waves).
// ---------------------------------------------------------------------------
__global__ void k_attn_top(const bf16_t* __restrict__ qg,
                           const bf16_t* __restrict__ kg,
                           const bf16_t* __restrict__ vT,
                           const int* __restrict__ topidx,
                           bf16_t* __restrict__ ctx) {
  extern __shared__ char smem[];
  float*  sc  = (float*)smem;              // [16][1024] scores -> P
  float*  red = sc + 16 * N_;              // [256]
  bf16_t* qs  = (bf16_t*)(red + 256);      // [16][64]
  __shared__ int tix[16];

  int bh   = blockIdx.x;
  int tile = blockIdx.y;
  int lane = threadIdx.x & 31, wave = threadIdx.x >> 5;
  int colbase = lane & 15, rowoff = (lane >> 4) << 3;

  if (threadIdx.x < 16) tix[threadIdx.x] = topidx[bh * U_ + tile * 16 + threadIdx.x];
  __syncthreads();

  { // gather the 16 selected q rows into LDS
    int r = threadIdx.x >> 3, seg = (threadIdx.x & 7) << 3;
    const bf16_t* src = qg + (bh * N_ + tix[r]) * E_ + seg;
#pragma unroll
    for (int i = 0; i < 8; ++i) qs[r * E_ + seg + i] = src[i];
  }
  __syncthreads();

  v16bf a0, a1;
  {
    int r = lane & 15, kb = (lane >> 4) << 3;
#pragma unroll
    for (int i = 0; i < 8; ++i) {
      a0[i]     = qs[r * E_ + kb + i];
      a0[i + 8] = qs[r * E_ + kb + 16 + i];
      a1[i]     = qs[r * E_ + 32 + kb + i];
      a1[i + 8] = qs[r * E_ + 32 + kb + 16 + i];
    }
  }
  // scores (scaled by 1/sqrt(64))
  for (int t = 0; t < 16; ++t) {
    int kr = (wave << 8) + (t << 4);
    const bf16_t* kbase = kg + (bh * N_ + kr) * E_;
    v8f acc = {};
    acc = wmma_bf16(a0, load_tile_g(kbase,      E_, lane), acc);
    acc = wmma_bf16(a1, load_tile_g(kbase + 32, E_, lane), acc);
#pragma unroll
    for (int i = 0; i < 8; ++i)
      sc[(rowoff + i) * N_ + kr + colbase] = acc[i] * 0.125f;
  }
  __syncthreads();

  // softmax per row (8 threads/row)
  int row = threadIdx.x >> 3, sub = threadIdx.x & 7;
  float mx = -3.4e38f;
  for (int j = sub; j < N_; j += 8) mx = fmaxf(mx, sc[row * N_ + j]);
  red[row * 8 + sub] = mx;
  __syncthreads();
  float rmax = red[row * 8];
  for (int j = 1; j < 8; ++j) rmax = fmaxf(rmax, red[row * 8 + j]);
  float se = 0.f;
  for (int j = sub; j < N_; j += 8) se += __expf(sc[row * N_ + j] - rmax);
  __syncthreads();
  red[row * 8 + sub] = se;
  __syncthreads();
  float denom = 0.f;
  for (int j = 0; j < 8; ++j) denom += red[row * 8 + j];
  float inv = 1.f / denom;
  for (int j = sub; j < N_; j += 8)
    sc[row * N_ + j] = __expf(sc[row * N_ + j] - rmax) * inv;
  __syncthreads();

  // P[16,1024] @ (vT[64,1024])^T, wave w -> e-columns [16w,16w+16)
  v8f o = {};
  for (int kk = 0; kk < N_; kk += 32) {
    v16bf pa;
    int r = lane & 15, kb = kk + ((lane >> 4) << 3);
#pragma unroll
    for (int i = 0; i < 8; ++i) {
      pa[i]     = f2bf(sc[r * N_ + kb + i]);
      pa[i + 8] = f2bf(sc[r * N_ + kb + 16 + i]);
    }
    const bf16_t* vb = vT + (bh * E_ + (wave << 4)) * N_ + kk;
    o = wmma_bf16(pa, load_tile_g(vb, N_, lane), o);
  }
  int h_ = bh & (H_ - 1), b_ = bh >> 3;
#pragma unroll
  for (int i = 0; i < 8; ++i) {
    int n_ = tix[rowoff + i];
    ctx[(b_ * N_ + n_) * DM_ + h_ * E_ + (wave << 4) + colbase] = f2bf(o[i]);
  }
}

// ---------------------------------------------------------------------------
// Kernel: GEMM (Nc=512) + bias + residual + LayerNorm.  One block owns 16
// full rows; 8 waves cover the 512 columns; A panel TDM-staged in LDS;
// LN over the row strip in LDS.
// ---------------------------------------------------------------------------
__global__ void k_gemm_ln(const bf16_t* __restrict__ A,
                          const bf16_t* __restrict__ W,
                          const float* __restrict__ bias,
                          const float* __restrict__ resid,
                          const float* __restrict__ g,
                          const float* __restrict__ beta,
                          int K, float* __restrict__ out_f,
                          bf16_t* __restrict__ out_b) {
  extern __shared__ char smem[];
  bf16_t* shA = (bf16_t*)smem;                   // [16][K]
  float*  o   = (float*)(smem + (size_t)16 * K * 2);  // [16][512]
  float*  red = o + 16 * DM_;                    // [512]
  int m0 = blockIdx.x << 4;
  int lane = threadIdx.x & 31, wave = threadIdx.x >> 5;
  int c0 = wave << 6;

  stage_panel(A + m0 * K, shA, 16, K, wave);

  v8f acc[4] = {};
  for (int kk = 0; kk < K; kk += 32) {
#pragma unroll
    for (int c = 0; c < 4; ++c)
      __builtin_prefetch((const void*)(W + (c0 + (c << 4)) * K + kk + 32), 0, 1);
    v16bf a = load_tile_g(shA + kk, K, lane);
#pragma unroll
    for (int c = 0; c < 4; ++c)
      acc[c] = wmma_bf16(a, load_tile_g(W + (c0 + (c << 4)) * K + kk, K, lane),
                         acc[c]);
  }
  int colbase = lane & 15, rowoff = (lane >> 4) << 3;
#pragma unroll
  for (int c = 0; c < 4; ++c)
#pragma unroll
    for (int i = 0; i < 8; ++i)
      o[(rowoff + i) * DM_ + c0 + (c << 4) + colbase] = acc[c][i];
  __syncthreads();

  int row = threadIdx.x >> 4, sub = threadIdx.x & 15;
  float sum = 0.f, ss = 0.f;
  for (int c = sub; c < DM_; c += 16) {
    float v = o[row * DM_ + c] + bias[c] + resid[(m0 + row) * DM_ + c];
    o[row * DM_ + c] = v;
    sum += v; ss += v * v;
  }
  red[row * 16 + sub]       = sum;
  red[256 + row * 16 + sub] = ss;
  __syncthreads();
  float tsum = 0.f, tss = 0.f;
  for (int j = 0; j < 16; ++j) {
    tsum += red[row * 16 + j];
    tss  += red[256 + row * 16 + j];
  }
  float mean = tsum * (1.f / (float)DM_);
  float var  = tss * (1.f / (float)DM_) - mean * mean;
  float rs   = rsqrtf(var + 1e-5f);
  for (int c = sub; c < DM_; c += 16) {
    float y = (o[row * DM_ + c] - mean) * rs * g[c] + beta[c];
    out_f[(m0 + row) * DM_ + c] = y;
    if (out_b) out_b[(m0 + row) * DM_ + c] = f2bf(y);
  }
}

// ---------------------------------------------------------------------------
extern "C" void kernel_launch(void* const* d_in, const int* in_sizes, int n_in,
                              void* d_out, int out_size, void* d_ws, size_t ws_size,
                              hipStream_t stream) {
  (void)in_sizes; (void)n_in; (void)out_size; (void)ws_size;
  const float* x   = (const float*)d_in[0];
  const int*   idx = (const int*)d_in[1];
  // d_in[2] = complexity (25) -> u = 256, baked into grid sizing
  const float* Wq = (const float*)d_in[3];  const float* bq  = (const float*)d_in[4];
  const float* Wk = (const float*)d_in[5];  const float* bk  = (const float*)d_in[6];
  const float* Wv = (const float*)d_in[7];  const float* bv  = (const float*)d_in[8];
  const float* Wo = (const float*)d_in[9];  const float* bo  = (const float*)d_in[10];
  const float* W1 = (const float*)d_in[11]; const float* b1  = (const float*)d_in[12];
  const float* W2 = (const float*)d_in[13]; const float* b2  = (const float*)d_in[14];
  const float* g1 = (const float*)d_in[15]; const float* be1 = (const float*)d_in[16];
  const float* g2 = (const float*)d_in[17]; const float* be2 = (const float*)d_in[18];
  float* out = (float*)d_out;

  char* wp = (char*)d_ws;
  auto carve = [&](size_t bytes) -> void* {
    void* p = (void*)wp;
    wp += (bytes + 255) & ~(size_t)255;
    return p;
  };
  bf16_t* x_bf  = (bf16_t*)carve((size_t)MR_ * DM_ * 2);
  bf16_t* Wq_bf = (bf16_t*)carve((size_t)DM_ * DM_ * 2);
  bf16_t* Wk_bf = (bf16_t*)carve((size_t)DM_ * DM_ * 2);
  bf16_t* Wv_bf = (bf16_t*)carve((size_t)DM_ * DM_ * 2);
  bf16_t* Wo_bf = (bf16_t*)carve((size_t)DM_ * DM_ * 2);
  bf16_t* W1_bf = (bf16_t*)carve((size_t)DFF_ * DM_ * 2);
  bf16_t* W2_bf = (bf16_t*)carve((size_t)DM_ * DFF_ * 2);
  bf16_t* q_bf  = (bf16_t*)carve((size_t)BH_ * N_ * E_ * 2);
  bf16_t* k_bf  = (bf16_t*)carve((size_t)BH_ * N_ * E_ * 2);
  bf16_t* vT_bf = (bf16_t*)carve((size_t)BH_ * E_ * N_ * 2);
  float*  Mm    = (float*) carve((size_t)BH_ * N_ * 4);
  int*    tixp  = (int*)   carve((size_t)BH_ * U_ * 4);
  float*  vmean = (float*) carve((size_t)BH_ * E_ * 4);
  bf16_t* ctx   = (bf16_t*)carve((size_t)MR_ * DM_ * 2);
  float*  x1_f  = (float*) carve((size_t)MR_ * DM_ * 4);
  bf16_t* x1_bf = (bf16_t*)carve((size_t)MR_ * DM_ * 2);
  bf16_t* h_bf  = (bf16_t*)carve((size_t)MR_ * DFF_ * 2);

  auto conv = [&](const float* s_, bf16_t* d_, int n) {
    k_f32_to_bf16<<<dim3((n + 255) / 256), dim3(256), 0, stream>>>(s_, d_, n);
  };
  conv(x,  x_bf,  MR_ * DM_);
  conv(Wq, Wq_bf, DM_ * DM_);
  conv(Wk, Wk_bf, DM_ * DM_);
  conv(Wv, Wv_bf, DM_ * DM_);
  conv(Wo, Wo_bf, DM_ * DM_);
  conv(W1, W1_bf, DFF_ * DM_);
  conv(W2, W2_bf, DM_ * DFF_);

  // QKV projections (strip GEMM, 8 waves/block, TDM-staged A panel: 16KB LDS)
  size_t shm_strip = (size_t)16 * DM_ * 2;
  k_gemm_strip<<<dim3(512), dim3(256), shm_strip, stream>>>(x_bf, Wq_bf, bq, MR_, DM_, DM_, 0, q_bf);
  k_gemm_strip<<<dim3(512), dim3(256), shm_strip, stream>>>(x_bf, Wk_bf, bk, MR_, DM_, DM_, 0, k_bf);
  k_gemm_strip<<<dim3(512), dim3(256), shm_strip, stream>>>(x_bf, Wv_bf, bv, MR_, DM_, DM_, 2, vT_bf);

  // sparsity measure M
  size_t shm_sc = (size_t)(16 * N_ + 256) * 4;
  k_scores_M<<<dim3(BH_, N_ / 16), dim3(128), shm_sc, stream>>>(q_bf, k_bf, idx, Mm);

  // stable top-256 per (b,h)
  k_topk<<<dim3(BH_), dim3(256), 0, stream>>>(Mm, tixp);

  // context init = mean of V
  k_vmean<<<dim3(BH_), dim3(E_), 0, stream>>>(vT_bf, vmean);
  k_ctx_init<<<dim3((MR_ * DM_ + 255) / 256), dim3(256), 0, stream>>>(vmean, ctx);

  // attention for selected queries
  size_t shm_at = (size_t)(16 * N_ + 256) * 4 + (size_t)16 * E_ * 2;
  k_attn_top<<<dim3(BH_, U_ / 16), dim3(128), shm_at, stream>>>(q_bf, k_bf, vT_bf, tixp, ctx);

  // Wo projection + residual + LN1 -> x1 (f32 + bf16)
  size_t shm_ln1 = (size_t)16 * DM_ * 2 + (size_t)(16 * DM_ + 512) * 4;
  k_gemm_ln<<<dim3(MR_ / 16), dim3(256), shm_ln1, stream>>>(
      ctx, Wo_bf, bo, x, g1, be1, DM_, x1_f, x1_bf);

  // FFN1: relu(x1 @ W1^T + b1): 16384 strips -> 2048 blocks
  k_gemm_strip<<<dim3(2048), dim3(256), shm_strip, stream>>>(x1_bf, W1_bf, b1, MR_, DFF_, DM_, 3, h_bf);

  // FFN2 + residual + LN2 -> d_out
  size_t shm_ln2 = (size_t)16 * DFF_ * 2 + (size_t)(16 * DM_ + 512) * 4;
  k_gemm_ln<<<dim3(MR_ / 16), dim3(256), shm_ln2, stream>>>(
      h_bf, W2_bf, b2, x1_f, g2, be2, DFF_, out, (bf16_t*)nullptr);
}
